// DiffAttn_57208964383284
// MI455X (gfx1250) — matrix operand order repaired and checked by reading, not compiled, not run on hardware
//
#include <hip/hip_runtime.h>

// ---------------------------------------------------------------------------
// Types for CDNA5 WMMA (wave32): A/B = 16 bf16 per lane, C/D = 8 f32 per lane.
// ---------------------------------------------------------------------------
typedef __attribute__((ext_vector_type(2)))  unsigned short v2u16;
typedef __attribute__((ext_vector_type(4)))  unsigned short v4u16;
typedef __attribute__((ext_vector_type(8)))  unsigned short v8u16;
typedef __attribute__((ext_vector_type(16))) unsigned short v16u16;
typedef __attribute__((ext_vector_type(16))) __bf16         v16bf;
typedef __attribute__((ext_vector_type(8)))  float          v8f;

__device__ __forceinline__ unsigned short f2bf(float f) {
  unsigned int u = __float_as_uint(f);
  u += 0x7FFFu + ((u >> 16) & 1u);          // round-to-nearest-even
  return (unsigned short)(u >> 16);
}

__device__ __forceinline__ v16bf mk16(v8u16 a, v8u16 b) {
  v16u16 r = __builtin_shufflevector(a, b, 0,1,2,3,4,5,6,7,8,9,10,11,12,13,14,15);
  return __builtin_bit_cast(v16bf, r);
}

// A-matrix fragment (16x32 bf16). Lane holds row (lane&15); within the 32-wide
// k-chunk starting at d0 the lane needs 16 contiguous k values starting at
// base = (lane>=16 ? 16 : 0)   (matches load_bfrag k order).
__device__ __forceinline__ v16bf load_afrag(const unsigned short* rowptr, int d0, int lane) {
  int off = d0 + ((lane >> 4) << 4);
  v8u16 x = *(const v8u16*)(rowptr + off);
  v8u16 y = *(const v8u16*)(rowptr + off + 8);
  return mk16(x, y);
}

// B-matrix fragment (32x16 bf16) from a row-major source whose rows are the
// N dimension: lane owns column (lane&15) and 16 contiguous k values starting
// at (lane>=16 ? 16 : 0)   (contiguous k order, matches load_afrag).
__device__ __forceinline__ v16bf load_bfrag(const unsigned short* rowptr, int lane) {
  int kb = (lane >> 4) << 4;
  v8u16 x = *(const v8u16*)(rowptr + kb);
  v8u16 y = *(const v8u16*)(rowptr + kb + 8);
  return mk16(x, y);
}

__device__ __forceinline__ v8f wmma_bf16(v16bf a, v16bf b, v8f c) {
  return __builtin_amdgcn_wmma_f32_16x16x32_bf16(false, a, false, b, (short)0, c, false, false);
}

// ---------------------------------------------------------------------------
// Kernel 1: f32 -> bf16 conversion (8 elements / thread, vectorized)
// ---------------------------------------------------------------------------
__global__ __launch_bounds__(256) void conv_bf16_kernel(const float* __restrict__ src,
                                                        unsigned short* __restrict__ dst) {
  size_t i = ((size_t)blockIdx.x * 256u + threadIdx.x) * 8u;
  v8f x = *(const v8f*)(src + i);
  v8u16 o;
#pragma unroll
  for (int e = 0; e < 8; ++e) o[e] = f2bf(x[e]);
  *(v8u16*)(dst + i) = o;
}

// ---------------------------------------------------------------------------
// Kernel 2: bf16 WMMA GEMM with fused bias, bf16 output.
//   C[M,N] = A[M,K] * W[K,N] + bias[N]
// Block = 256 threads = 8 waves, tile 128x128, K-step 32, register
// double-buffered global->LDS staging. Wave grid 4(m) x 2(n); each wave
// computes a 32x64 tile = 8 accumulators, 8 WMMA per k-step.
// ---------------------------------------------------------------------------
__global__ __launch_bounds__(256) void gemm_bias_bf16(const unsigned short* __restrict__ A,
                                                      const unsigned short* __restrict__ W,
                                                      const float* __restrict__ bias,
                                                      unsigned short* __restrict__ C,
                                                      int M, int N, int K) {
  __shared__ unsigned short As[128][32];   // A tile, row-major              8 KB
  __shared__ unsigned short Bt[128][32];   // W tile transposed: Bt[n][k]    8 KB

  const int t = threadIdx.x, lane = t & 31, wave = t >> 5;
  const int wm = wave >> 1, wn = wave & 1;
  const int hi = lane >> 4, nl = lane & 15;
  const int mbase = blockIdx.y * 128, nbase = blockIdx.x * 128;

  // staging thread mapping
  const int ar = t >> 1, ac = (t & 1) * 16;        // A: 128 rows x 32 cols
  const int wk = (t >> 4) * 2, wc = (t & 15) * 8;  // W: 32 rows  x 128 cols (2 rows/thread)

  v8f acc[2][4] = {};
  v8u16 ra0, ra1, rb0, rb1;

  // prologue: fetch tile k0=0 into registers
  ra0 = *(const v8u16*)(A + (size_t)(mbase + ar) * K + ac);
  ra1 = *(const v8u16*)(A + (size_t)(mbase + ar) * K + ac + 8);
  rb0 = *(const v8u16*)(W + (size_t)wk * N + nbase + wc);
  rb1 = *(const v8u16*)(W + (size_t)(wk + 1) * N + nbase + wc);

  for (int k0 = 0; k0 < K; k0 += 32) {
    __syncthreads();                       // previous tile fully consumed
    *(v8u16*)&As[ar][ac]     = ra0;
    *(v8u16*)&As[ar][ac + 8] = ra1;
#pragma unroll
    for (int e = 0; e < 8; ++e)
      *(v2u16*)&Bt[wc + e][wk] = (v2u16){rb0[e], rb1[e]};
    __syncthreads();

    if (k0 + 32 < K) {                     // overlap next-tile fetch with WMMA
      int k1 = k0 + 32;
      ra0 = *(const v8u16*)(A + (size_t)(mbase + ar) * K + k1 + ac);
      ra1 = *(const v8u16*)(A + (size_t)(mbase + ar) * K + k1 + ac + 8);
      rb0 = *(const v8u16*)(W + (size_t)(k1 + wk) * N + nbase + wc);
      rb1 = *(const v8u16*)(W + (size_t)(k1 + wk + 1) * N + nbase + wc);
    }

    v16bf af0 = load_afrag(&As[wm * 32 + nl][0], 0, lane);
    v16bf af1 = load_afrag(&As[wm * 32 + 16 + nl][0], 0, lane);
#pragma unroll
    for (int ni = 0; ni < 4; ++ni) {
      v16bf bf = load_bfrag(&Bt[wn * 64 + ni * 16 + nl][0], lane);
      acc[0][ni] = wmma_bf16(af0, bf, acc[0][ni]);
      acc[1][ni] = wmma_bf16(af1, bf, acc[1][ni]);
    }
  }

  // Epilogue: C/D layout -> n = lane&15, m = j + (lane>>4)*8
#pragma unroll
  for (int mi = 0; mi < 2; ++mi)
#pragma unroll
    for (int ni = 0; ni < 4; ++ni)
#pragma unroll
      for (int j = 0; j < 8; ++j) {
        int m = mbase + wm * 32 + mi * 16 + hi * 8 + j;
        int n = nbase + wn * 64 + ni * 16 + nl;
        C[(size_t)m * N + n] = f2bf(acc[mi][ni][j] + bias[n]);
      }
}

// ---------------------------------------------------------------------------
// Kernel 3: fused differential flash attention (bf16 WMMA, f32 accumulate).
// Block = 256 threads = 8 waves handles 16 queries x full D=512.
// Wave w owns D-slice [w*64, w*64+64): Q frags in registers, two 16x64 f32
// output accumulators (branch 1 / branch 2). Key tiles of 32; score partials
// over D-slices reduced through LDS; dual online softmax; P@V via WMMA with
// V staged transposed in LDS (packed b64 stores). Final: O1/l1 - lam * O2/l2.
// ---------------------------------------------------------------------------
__global__ __launch_bounds__(256) void diff_attn_kernel(const unsigned short* __restrict__ Qg,
                                                        const unsigned short* __restrict__ Kg,
                                                        const unsigned short* __restrict__ Vg,
                                                        const float* __restrict__ lamp,
                                                        float* __restrict__ out) {
  constexpr int S = 2048, D = 512, QD = 1024;
  constexpr float sscale = 0.04419417382415922f;   // 1/sqrt(512)

  __shared__ unsigned short Vt[512][32];           // V tile transposed: Vt[d][key]  32 KB
  __shared__ float SP[8][16][32];                  // per-wave score partials        16 KB
  __shared__ unsigned short Pb[2][16][32];         // exp-probabilities (bf16)        2 KB
  __shared__ float rowM[2][16], rowL[2][16], rowSc[2][16];

  const int b = blockIdx.y;
  const int q0 = blockIdx.x * 16;
  const int t = threadIdx.x, lane = t & 31, wave = t >> 5;
  const int dbase = wave * 64;
  const int hi = lane >> 4, nl = lane & 15;
  const size_t bS = (size_t)b * S;

  // Preload Q fragments for this wave's D-slice: [branch][k-chunk]
  const unsigned short* Qrow = Qg + (bS + q0 + nl) * QD;
  v16bf qf[2][2];
#pragma unroll
  for (int br = 0; br < 2; ++br)
#pragma unroll
    for (int kc = 0; kc < 2; ++kc)
      qf[br][kc] = load_afrag(Qrow, br * D + dbase + kc * 32, lane);

  if (t < 32) { rowM[t >> 4][t & 15] = -1e30f; rowL[t >> 4][t & 15] = 0.0f; }

  v8f o1[4] = {}; v8f o2[4] = {};

  for (int kt = 0; kt < S; kt += 32) {
    __syncthreads();   // previous tile fully consumed: safe to restage Vt / SP

    // ---- stage V tile transposed: Vt[d][key]; each thread packs 4 keys x 16 d
    {
      int r4 = (t & 7) * 4;          // 4 consecutive keys
      int dc = (t >> 3) * 16;        // 16 d-values
      const unsigned short* vp = Vg + (bS + kt + r4) * D + dc;
      v8u16 x0a = *(const v8u16*)(vp);
      v8u16 x0b = *(const v8u16*)(vp + 8);
      v8u16 x1a = *(const v8u16*)(vp + D);
      v8u16 x1b = *(const v8u16*)(vp + D + 8);
      v8u16 x2a = *(const v8u16*)(vp + 2 * D);
      v8u16 x2b = *(const v8u16*)(vp + 2 * D + 8);
      v8u16 x3a = *(const v8u16*)(vp + 3 * D);
      v8u16 x3b = *(const v8u16*)(vp + 3 * D + 8);
#pragma unroll
      for (int e = 0; e < 8; ++e) {
        *(v4u16*)&Vt[dc + e][r4]     = (v4u16){x0a[e], x1a[e], x2a[e], x3a[e]};
        *(v4u16*)&Vt[dc + 8 + e][r4] = (v4u16){x0b[e], x1b[e], x2b[e], x3b[e]};
      }
    }

    if (kt + 32 < S)   // gfx1250 global_prefetch_b8 for next key tile
      __builtin_prefetch(Kg + (bS + kt + 32 + nl) * QD + dbase, 0, 1);

    // ---- per-branch: partial scores (this wave's D-slice) + reduce + softmax
#pragma unroll
    for (int br = 0; br < 2; ++br) {
      v8f s0 = {}; v8f s1 = {};
#pragma unroll
      for (int kc = 0; kc < 2; ++kc) {
        v16bf a = qf[br][kc];
        const unsigned short* k0p = Kg + (bS + kt + nl) * QD + br * D + dbase + kc * 32;
        s0 = wmma_bf16(a, load_bfrag(k0p, lane), s0);
        s1 = wmma_bf16(a, load_bfrag(k0p + 16 * QD, lane), s1);
      }
#pragma unroll
      for (int j = 0; j < 8; ++j) {
        SP[wave][hi * 8 + j][nl]      = s0[j];
        SP[wave][hi * 8 + j][16 + nl] = s1[j];
      }
      __syncthreads();

      // reduce 8 wave-partials + dual online softmax update (16 lanes / row)
      {
        int row = t >> 4;
        int c0  = (t & 15) * 2;
        float v0 = 0.0f, v1 = 0.0f;
#pragma unroll
        for (int w = 0; w < 8; ++w) { v0 += SP[w][row][c0]; v1 += SP[w][row][c0 + 1]; }
        v0 *= sscale; v1 *= sscale;
        float mx = fmaxf(v0, v1);
        mx = fmaxf(mx, __shfl_xor(mx, 1, 32));
        mx = fmaxf(mx, __shfl_xor(mx, 2, 32));
        mx = fmaxf(mx, __shfl_xor(mx, 4, 32));
        mx = fmaxf(mx, __shfl_xor(mx, 8, 32));
        float mo = rowM[br][row];
        float mn = fmaxf(mo, mx);
        float sc = __expf(mo - mn);
        float p0 = __expf(v0 - mn), p1 = __expf(v1 - mn);
        Pb[br][row][c0]     = f2bf(p0);
        Pb[br][row][c0 + 1] = f2bf(p1);
        float ls = p0 + p1;
        ls += __shfl_xor(ls, 1, 32);
        ls += __shfl_xor(ls, 2, 32);
        ls += __shfl_xor(ls, 4, 32);
        ls += __shfl_xor(ls, 8, 32);
        if ((t & 15) == 0) {
          rowM[br][row]  = mn;
          rowL[br][row]  = rowL[br][row] * sc + ls;
          rowSc[br][row] = sc;
        }
      }
      __syncthreads();
    }

    // ---- rescale accumulators, then P@V for this wave's D-slice ----
#pragma unroll
    for (int j = 0; j < 8; ++j) {
      int row = hi * 8 + j;
      float s1v = rowSc[0][row], s2v = rowSc[1][row];
#pragma unroll
      for (int cc = 0; cc < 4; ++cc) { o1[cc][j] *= s1v; o2[cc][j] *= s2v; }
    }
    v16bf p1 = load_afrag(&Pb[0][nl][0], 0, lane);
    v16bf p2 = load_afrag(&Pb[1][nl][0], 0, lane);
#pragma unroll
    for (int cc = 0; cc < 4; ++cc) {
      v16bf vb = load_bfrag(&Vt[dbase + cc * 16 + nl][0], lane);
      o1[cc] = wmma_bf16(p1, vb, o1[cc]);
      o2[cc] = wmma_bf16(p2, vb, o2[cc]);
    }
  }

  __syncthreads();
  const float lam = lamp[0];
#pragma unroll
  for (int j = 0; j < 8; ++j) {
    int row = hi * 8 + j;
    float r1 = 1.0f / rowL[0][row];
    float r2 = 1.0f / rowL[1][row];
#pragma unroll
    for (int cc = 0; cc < 4; ++cc) {
      int col = dbase + cc * 16 + nl;
      out[(bS + q0 + row) * D + col] = o1[cc][j] * r1 - lam * o2[cc][j] * r2;
    }
  }
}

// ---------------------------------------------------------------------------
// Host launcher
// ---------------------------------------------------------------------------
extern "C" void kernel_launch(void* const* d_in, const int* in_sizes, int n_in,
                              void* d_out, int out_size, void* d_ws, size_t ws_size,
                              hipStream_t stream) {
  (void)in_sizes; (void)n_in; (void)out_size; (void)ws_size;
  constexpr int B = 8, S = 2048, E = 1024, D = 512;
  constexpr int M = B * S;                         // 16384 query/key rows

  const float* X   = (const float*)d_in[0];
  const float* lam = (const float*)d_in[1];
  const float* Wq  = (const float*)d_in[2];
  const float* bq  = (const float*)d_in[3];
  const float* Wk  = (const float*)d_in[4];
  const float* bk  = (const float*)d_in[5];
  const float* Wv  = (const float*)d_in[6];
  const float* bv  = (const float*)d_in[7];
  float* out = (float*)d_out;

  // Workspace partition (bf16 = unsigned short payloads)
  unsigned short* Xbf = (unsigned short*)d_ws;          // M*E
  unsigned short* Wqb = Xbf + (size_t)M * E;            // E*2D
  unsigned short* Wkb = Wqb + (size_t)E * 2 * D;        // E*2D
  unsigned short* Wvb = Wkb + (size_t)E * 2 * D;        // E*D
  unsigned short* Qb  = Wvb + (size_t)E * D;            // M*2D
  unsigned short* Kb  = Qb  + (size_t)M * 2 * D;        // M*2D
  unsigned short* Vb  = Kb  + (size_t)M * 2 * D;        // M*D

  // 1) f32 -> bf16 conversions (all sizes divisible by 2048)
  conv_bf16_kernel<<<(size_t)M * E / 2048, 256, 0, stream>>>(X,  Xbf);
  conv_bf16_kernel<<<(size_t)E * 2 * D / 2048, 256, 0, stream>>>(Wq, Wqb);
  conv_bf16_kernel<<<(size_t)E * 2 * D / 2048, 256, 0, stream>>>(Wk, Wkb);
  conv_bf16_kernel<<<(size_t)E * D / 2048, 256, 0, stream>>>(Wv, Wvb);

  // 2) projections: Q, K (M x 1024), V (M x 512), bf16 out with fused bias
  gemm_bias_bf16<<<dim3(2 * D / 128, M / 128), 256, 0, stream>>>(Xbf, Wqb, bq, Qb, M, 2 * D, E);
  gemm_bias_bf16<<<dim3(2 * D / 128, M / 128), 256, 0, stream>>>(Xbf, Wkb, bk, Kb, M, 2 * D, E);
  gemm_bias_bf16<<<dim3(D / 128,     M / 128), 256, 0, stream>>>(Xbf, Wvb, bv, Vb, M, D, E);

  // 3) fused differential flash attention
  diff_attn_kernel<<<dim3(S / 16, B), 256, 0, stream>>>(Qb, Kb, Vb, lam, out);
}
